// MultiHeadAttention_8443905704287
// MI455X (gfx1250) — matrix operand order repaired and checked by reading, not compiled, not run on hardware
//
#include <hip/hip_runtime.h>

// ---------------------------------------------------------------------------
// MI455X (gfx1250) fused multi-head-over-heads attention.
//   B=8, S=4096, E=1024, H=16, HD=64, T=B*S=32768 tokens.
// Matrix math via V_WMMA_F32_16X16X32_BF16 (bf16 in, f32 accumulate).
// Output GEMM uses GLOBAL_LOAD_ASYNC_TO_LDS_B128 double buffering when the
// toolchain exposes the builtin (guarded by __has_builtin).
// ---------------------------------------------------------------------------

typedef __attribute__((ext_vector_type(16))) __bf16 v16bf;
typedef __attribute__((ext_vector_type(8)))  float  v8f;

#define T_TOKENS   32768
#define E_DIM      1024
#define H_HEADS    16
#define HD_DIM     64
#define SCALE_F    0.125f   // 1/sqrt(64)

__device__ __forceinline__ unsigned short f2bf(float f) {
  unsigned u = __float_as_uint(f);
  u += 0x7FFFu + ((u >> 16) & 1u);          // round-to-nearest-even
  return (unsigned short)(u >> 16);
}

union BFrag {                // one WMMA bf16 A/B operand (8 VGPRs / 32 bytes)
  v16bf          v;
  unsigned short u[16];
  uint4          q[2];
};

__device__ __forceinline__ v8f wmma_bf16(const v16bf& a, const v16bf& b, const v8f& c) {
  return __builtin_amdgcn_wmma_f32_16x16x32_bf16(false, a, false, b, (short)0, c, false, false);
}

// ---- optional CDNA5 async global->LDS path -------------------------------
#if defined(__has_builtin)
#if __has_builtin(__builtin_amdgcn_global_load_async_to_lds_b128)
#define HAVE_ASYNC_LDS 1
#endif
#endif

#ifdef HAVE_ASYNC_LDS
// Builtin signature (from compiler diagnostic): param0 is a pointer to a
// GCC-vector 'int __attribute__((vector_size(16)))' in the global (AS1)
// address space; param1 analogous in LDS (AS3); then imm offset, imm cpol.
typedef int gvec4i __attribute__((vector_size(16)));

__device__ __forceinline__ void async_cp_b128(const void* g, void* l) {
  // generic LDS va low 32 bits are the LDS byte offset (ISA aperture rule),
  // so integer truncation yields a valid AS(3) pointer value.
  __builtin_amdgcn_global_load_async_to_lds_b128(
      (__attribute__((address_space(1))) gvec4i*)(unsigned long long)g,
      (__attribute__((address_space(3))) gvec4i*)(unsigned int)(unsigned long long)l,
      0, 0);
}
__device__ __forceinline__ void async_wait0() {
#if __has_builtin(__builtin_amdgcn_s_wait_asynccnt)
  __builtin_amdgcn_s_wait_asynccnt(0);
#else
  asm volatile("s_wait_asynccnt 0x0" ::: "memory");
#endif
}
#endif

// ---------------------------------------------------------------------------
// Kernel 0: fp32 -> bf16 weight conversion into workspace.
// ---------------------------------------------------------------------------
__global__ void convert_kernel(const float* __restrict__ wo, const float* __restrict__ wq,
                               const float* __restrict__ wk, const float* __restrict__ wv,
                               unsigned short* __restrict__ wo_bf, unsigned short* __restrict__ wq_bf,
                               unsigned short* __restrict__ wk_bf, unsigned short* __restrict__ wv_bf) {
  int idx = blockIdx.x * blockDim.x + threadIdx.x;
  if (idx < E_DIM * E_DIM) wo_bf[idx] = f2bf(wo[idx]);
  if (idx < HD_DIM * HD_DIM) {
    wq_bf[idx] = f2bf(wq[idx]);
    wk_bf[idx] = f2bf(wk[idx]);
    wv_bf[idx] = f2bf(wv[idx]);
  }
}

// ---------------------------------------------------------------------------
// Kernel 1: per-token QKV projection + heads-axis attention.
// One wave32 per token; 4 tokens (4 waves, 128 threads) per block.
// 30 v_wmma_f32_16x16x32_bf16 per token.
// ---------------------------------------------------------------------------
#define TOK_PER_BLK 4

__global__ __launch_bounds__(128) void qkv_attn_kernel(
    const float* __restrict__ x,
    const unsigned short* __restrict__ wq_bf,
    const unsigned short* __restrict__ wk_bf,
    const unsigned short* __restrict__ wv_bf,
    const float* __restrict__ bq,
    const float* __restrict__ bk,
    const float* __restrict__ bv,
    unsigned short* __restrict__ vals_bf)   // [T, 1024] bf16, flat index n*64+w
{
  __shared__ float          xs [TOK_PER_BLK][1024];      // staged x row
  __shared__ unsigned short qs [TOK_PER_BLK][16][72];    // q[head][w'], padded rows
  __shared__ unsigned short ks_[TOK_PER_BLK][16][72];    // k[head][w']
  __shared__ unsigned short vts[TOK_PER_BLK][64][24];    // v transposed [w'][head]
  __shared__ float          ls [TOK_PER_BLK][16][20];    // logits / attn [n][m]

  const int tid  = threadIdx.x;
  const int lane = tid & 31;
  const int wv   = tid >> 5;            // wave id == token-in-block
  const int half = lane >> 4;           // 0: lanes 0-15, 1: lanes 16-31
  const int m    = lane & 15;
  const long token = (long)blockIdx.x * TOK_PER_BLK + wv;

  // ---- stage x row into LDS (coalesced b128) ----
  {
    const float4* src = (const float4*)(x + (size_t)token * E_DIM);
    float4* dst = (float4*)xs[wv];
#pragma unroll
    for (int i = 0; i < 8; ++i) dst[lane + 32 * i] = src[lane + 32 * i];
  }
  __syncthreads();

  // ---- A fragments of xh (M=head=16, K=w=64): xh[n,w] = x[w*16+n] ----
  BFrag xa[2];
#pragma unroll
  for (int kt = 0; kt < 2; ++kt) {
#pragma unroll
    for (int i = 0; i < 16; ++i) {
      int kk = kt * 32 + half * 8 + (i < 8 ? i : i + 8);   // ISA 16-bit A layout
      xa[kt].u[i] = f2bf(xs[wv][kk * 16 + m]);
    }
  }

  // ---- q / k / v projections: [16x64] = xh @ W^T + b ----
#pragma unroll
  for (int p = 0; p < 3; ++p) {
    const unsigned short* W   = (p == 0) ? wq_bf : ((p == 1) ? wk_bf : wv_bf);
    const float*          bia = (p == 0) ? bq    : ((p == 1) ? bk    : bv);
#pragma unroll
    for (int nt = 0; nt < 4; ++nt) {
      const int n = nt * 16 + m;                  // output dim w' (N lane)
      const float bval = bia[n];
      v8f acc;
#pragma unroll
      for (int r = 0; r < 8; ++r) acc[r] = bval;  // bias folded into C
#pragma unroll
      for (int kt = 0; kt < 2; ++kt) {
        // B[k=w][n=w'] = W[n*64+k]; lane half picks K runs [kt*32+half*8, +8) and [kt*32+16+half*8, +8)
        BFrag bf;
        const uint4* p4 = (const uint4*)(W + n * 64 + kt * 32 + half * 8);
        bf.q[0] = p4[0];  bf.q[1] = p4[2];
        acc = wmma_bf16(xa[kt].v, bf.v, acc);
      }
      if (p == 2) {       // v stored transposed: vts[w'][head], packed b128 store
        uint4 packed;
        unsigned* pu = (unsigned*)&packed;
#pragma unroll
        for (int r = 0; r < 4; ++r)
          pu[r] = (unsigned)f2bf(acc[2 * r]) | ((unsigned)f2bf(acc[2 * r + 1]) << 16);
        *(uint4*)&vts[wv][n][half * 8] = packed;
      } else {
        unsigned short (*dst)[72] = (p == 0) ? qs[wv] : ks_[wv];
#pragma unroll
        for (int r = 0; r < 8; ++r) dst[r + half * 8][n] = f2bf(acc[r]);
      }
    }
  }
  __syncthreads();

  // ---- logits[n,m] = (q @ k^T) * SCALE : A=q (M=n,K=w'), B=k^T (K=w',N=m) ----
  {
    v8f acc;
#pragma unroll
    for (int r = 0; r < 8; ++r) acc[r] = 0.f;
#pragma unroll
    for (int kt = 0; kt < 2; ++kt) {
      BFrag qa, kb;
      qa.q[0] = *(const uint4*)&qs[wv][m][kt * 32 + half * 8];
      qa.q[1] = *(const uint4*)&qs[wv][m][kt * 32 + half * 8 + 16];
      kb.q[0] = *(const uint4*)&ks_[wv][m][kt * 32 + half * 8];
      kb.q[1] = *(const uint4*)&ks_[wv][m][kt * 32 + half * 8 + 16];
      acc = wmma_bf16(qa.v, kb.v, acc);
    }
#pragma unroll
    for (int r = 0; r < 8; ++r) ls[wv][r + half * 8][m] = acc[r] * SCALE_F;
  }
  __syncthreads();

  // ---- softmax over m (columns): lanes 0..15 own one row each ----
  if (lane < 16) {
    float row[16], vmax = -3.0e38f;
#pragma unroll
    for (int c = 0; c < 16; ++c) { row[c] = ls[wv][lane][c]; vmax = fmaxf(vmax, row[c]); }
    float s = 0.f;
#pragma unroll
    for (int c = 0; c < 16; ++c) { row[c] = __expf(row[c] - vmax); s += row[c]; }
    const float inv = 1.f / s;
#pragma unroll
    for (int c = 0; c < 16; ++c) ls[wv][lane][c] = row[c] * inv;
  }
  __syncthreads();

  // ---- vals = attn @ v : A=attn (16x16, K padded to 32 with zeros) ----
  BFrag aa;
  {
    const float4 a0 = *(const float4*)&ls[wv][m][half * 8];
    const float4 a1 = *(const float4*)&ls[wv][m][half * 8 + 4];
    aa.u[0] = f2bf(a0.x); aa.u[1] = f2bf(a0.y); aa.u[2] = f2bf(a0.z); aa.u[3] = f2bf(a0.w);
    aa.u[4] = f2bf(a1.x); aa.u[5] = f2bf(a1.y); aa.u[6] = f2bf(a1.z); aa.u[7] = f2bf(a1.w);
#pragma unroll
    for (int i = 8; i < 16; ++i) aa.u[i] = 0;   // K=16..31 pad
  }
#pragma unroll
  for (int nt = 0; nt < 4; ++nt) {
    BFrag vb;
    {                     // elements 0-7: K=half*8..+7 from vts; elements 8-15: K=16..31 -> zero pad
      vb.q[0] = *(const uint4*)&vts[wv][nt * 16 + m][half * 8];
      vb.q[1] = make_uint4(0u, 0u, 0u, 0u);
    }
    v8f acc;
#pragma unroll
    for (int r = 0; r < 8; ++r) acc[r] = 0.f;
    acc = wmma_bf16(aa.v, vb.v, acc);
    // write bf16 vals, flattened as head*64 + w'
    const size_t base = (size_t)token * E_DIM + nt * 16 + m;
#pragma unroll
    for (int r = 0; r < 8; ++r)
      vals_bf[base + (size_t)(r + half * 8) * HD_DIM] = f2bf(acc[r]);
  }
}

// ---------------------------------------------------------------------------
// Kernel 2: out = vals @ wo^T + bo   ([32768x1024] x [1024x1024], f32 out)
// 128x128 block tile, 8 waves (2x4), 64x32 per wave, K stepped by 32.
// Double-buffered LDS; async global->LDS copies when available.
// LDS rows padded to 40 bf16 (80 B) -> conflict-free b128 fragment reads.
// ---------------------------------------------------------------------------
__global__ __launch_bounds__(256) void out_proj_kernel(
    const unsigned short* __restrict__ vals_bf,
    const unsigned short* __restrict__ wo_bf,
    const float* __restrict__ bo,
    float* __restrict__ out)
{
  __shared__ unsigned short As[2][128][40];
  __shared__ unsigned short Bs[2][128][40];

  const int tid  = threadIdx.x;
  const int lane = tid & 31;
  const int wv   = tid >> 5;
  const int half = lane >> 4;
  const int m    = lane & 15;
  const int wm   = wv & 1;              // 2 wave-rows of 64
  const int wn   = wv >> 1;             // 4 wave-cols of 32
  const long m0  = (long)blockIdx.x * 128;
  const int  n0  = blockIdx.y * 128;

  const unsigned short* aBase = vals_bf + (size_t)m0 * E_DIM;
  const unsigned short* bBase = wo_bf + (size_t)n0 * E_DIM;

  // this thread's two 16-byte chunks of each 128x32 tile
  const int c0 = tid,        r0 = c0 >> 2, o0 = (c0 & 3) * 8;
  const int c1 = tid + 256,  r1 = c1 >> 2, o1 = (c1 & 3) * 8;

  v8f acc[4][2];
#pragma unroll
  for (int j = 0; j < 2; ++j) {
    const float bval = bo[n0 + wn * 32 + j * 16 + m];   // bias folded into C
#pragma unroll
    for (int i = 0; i < 4; ++i)
#pragma unroll
      for (int r = 0; r < 8; ++r) acc[i][j][r] = bval;
  }

  auto compute = [&](int buf) {
    BFrag af[4], bfr[2];
#pragma unroll
    for (int i = 0; i < 4; ++i) {
      const int row = wm * 64 + i * 16 + m;
      af[i].q[0] = *(const uint4*)&As[buf][row][half * 8];
      af[i].q[1] = *(const uint4*)&As[buf][row][half * 8 + 16];
    }
#pragma unroll
    for (int j = 0; j < 2; ++j) {
      const int row = wn * 32 + j * 16 + m;
      bfr[j].q[0] = *(const uint4*)&Bs[buf][row][half * 8];
      bfr[j].q[1] = *(const uint4*)&Bs[buf][row][half * 8 + 16];
    }
#pragma unroll
    for (int i = 0; i < 4; ++i)
#pragma unroll
      for (int j = 0; j < 2; ++j)
        acc[i][j] = wmma_bf16(af[i].v, bfr[j].v, acc[i][j]);
  };

#ifdef HAVE_ASYNC_LDS
  // ---- async double-buffered pipeline (GLOBAL_LOAD_ASYNC_TO_LDS_B128) ----
  auto issue = [&](int kk, int buf) {
    const int k0 = kk * 32;
    async_cp_b128(aBase + (size_t)r0 * E_DIM + k0 + o0, &As[buf][r0][o0]);
    async_cp_b128(aBase + (size_t)r1 * E_DIM + k0 + o1, &As[buf][r1][o1]);
    async_cp_b128(bBase + (size_t)r0 * E_DIM + k0 + o0, &Bs[buf][r0][o0]);
    async_cp_b128(bBase + (size_t)r1 * E_DIM + k0 + o1, &Bs[buf][r1][o1]);
  };
  issue(0, 0);
  async_wait0();
  __syncthreads();
  for (int kk = 0; kk < 32; ++kk) {
    const int cur = kk & 1;
    if (kk < 31) issue(kk + 1, cur ^ 1);   // prefetch next tile into other buffer
    compute(cur);                          // 8 WMMAs overlap the async copies
    async_wait0();
    __syncthreads();
  }
#else
  // ---- synchronous fallback: register-staged double buffering ----
  uint4 pa0, pa1, pb0, pb1;
  auto fetch = [&](int kk) {
    const int k0 = kk * 32;
    pa0 = *(const uint4*)(aBase + (size_t)r0 * E_DIM + k0 + o0);
    pa1 = *(const uint4*)(aBase + (size_t)r1 * E_DIM + k0 + o1);
    pb0 = *(const uint4*)(bBase + (size_t)r0 * E_DIM + k0 + o0);
    pb1 = *(const uint4*)(bBase + (size_t)r1 * E_DIM + k0 + o1);
  };
  auto commit = [&](int buf) {
    *(uint4*)&As[buf][r0][o0] = pa0;
    *(uint4*)&As[buf][r1][o1] = pa1;
    *(uint4*)&Bs[buf][r0][o0] = pb0;
    *(uint4*)&Bs[buf][r1][o1] = pb1;
  };
  fetch(0); commit(0);
  __syncthreads();
  for (int kk = 0; kk < 32; ++kk) {
    const int cur = kk & 1;
    if (kk < 31) fetch(kk + 1);
    compute(cur);
    if (kk < 31) commit(cur ^ 1);
    __syncthreads();
  }
#endif

  // epilogue: coalesced f32 stores (lanes 0-15 / 16-31 each contiguous)
#pragma unroll
  for (int i = 0; i < 4; ++i)
#pragma unroll
    for (int j = 0; j < 2; ++j)
#pragma unroll
      for (int r = 0; r < 8; ++r) {
        const long row = m0 + wm * 64 + i * 16 + r + half * 8;
        const int  col = n0 + wn * 32 + j * 16 + m;
        out[row * E_DIM + col] = acc[i][j][r];
      }
}

// ---------------------------------------------------------------------------
// Host launcher
// ---------------------------------------------------------------------------
extern "C" void kernel_launch(void* const* d_in, const int* in_sizes, int n_in,
                              void* d_out, int out_size, void* d_ws, size_t ws_size,
                              hipStream_t stream) {
  const float* x  = (const float*)d_in[0];
  const float* wq = (const float*)d_in[1];
  const float* bq = (const float*)d_in[2];
  const float* wk = (const float*)d_in[3];
  const float* bk = (const float*)d_in[4];
  const float* wv = (const float*)d_in[5];
  const float* bv = (const float*)d_in[6];
  const float* wo = (const float*)d_in[7];
  const float* bo = (const float*)d_in[8];
  float* out = (float*)d_out;

  // workspace layout (bf16): vals [32768*1024] | wo [1024*1024] | wq/wk/wv [4096 each]
  unsigned short* vals_bf = (unsigned short*)d_ws;
  unsigned short* wo_bf   = vals_bf + (size_t)T_TOKENS * E_DIM;
  unsigned short* wq_bf   = wo_bf + (size_t)E_DIM * E_DIM;
  unsigned short* wk_bf   = wq_bf + HD_DIM * HD_DIM;
  unsigned short* wv_bf   = wk_bf + HD_DIM * HD_DIM;

  convert_kernel<<<(E_DIM * E_DIM + 255) / 256, 256, 0, stream>>>(
      wo, wq, wk, wv, wo_bf, wq_bf, wk_bf, wv_bf);

  qkv_attn_kernel<<<T_TOKENS / TOK_PER_BLK, 128, 0, stream>>>(
      x, wq_bf, wk_bf, wv_bf, bq, bk, bv, vals_bf);

  out_proj_kernel<<<dim3(T_TOKENS / 128, E_DIM / 128), 256, 0, stream>>>(
      vals_bf, wo_bf, bo, out);
}
